// FlashAttention_24300924960810
// MI455X (gfx1250) — matrix operand order repaired and checked by reading, not compiled, not run on hardware
//
#include <hip/hip_runtime.h>
#include <hip/hip_bf16.h>

// Flash attention for B=2, S=2048, H=16, Dh=64, fp32 in/out.
// Transposed formulation: S^T = K * Q^T, O^T = V^T * P^T (all WMMA
// v_wmma_f32_16x16x32_f16). Each lane owns ONE query row in the softmax
// (S^T C-layout): row max needs a single shfl_xor(16); the row SUM is
// computed by the matrix pipe itself via a constant ones A-fragment
// (ones(16x32) x P^T = row sums), accumulated online exactly like O^T.

#define S_LEN   2048
#define DIMC    1024
#define NHEADS  16
#define DH      64
#define KB      64              // keys per block iteration
#define WAVES   4               // waves per workgroup (each owns 16 q rows)
#define KSTRIDE (DH + 8)        // K tile [key][d], padded halves (144B rows)
#define VSTRIDE (KB + 8)        // V tile transposed [d][key], padded halves
#define PSTRIDE (KB + 8)        // P^T tile [qrow][key], padded halves

typedef __attribute__((ext_vector_type(16))) _Float16 v16h;
typedef __attribute__((ext_vector_type(8)))  _Float16 v8h;
typedef __attribute__((ext_vector_type(8)))  float    v8f;

__global__ __launch_bounds__(128) void
fa_fwd_kernel(const float* __restrict__ Q,
              const float* __restrict__ K,
              const float* __restrict__ V,
              float* __restrict__ O)
{
    __shared__ __attribute__((aligned(16))) _Float16 Ks[KB * KSTRIDE];
    __shared__ __attribute__((aligned(16))) _Float16 Vt[DH * VSTRIDE];
    __shared__ __attribute__((aligned(16))) _Float16 Pt[WAVES * 16 * PSTRIDE];

    const int tid  = threadIdx.x;
    const int wave = tid >> 5;
    const int lane = tid & 31;
    const int half = lane >> 4;   // 16-lane half of the wave
    const int ln   = lane & 15;

    // block -> (b, h, q-block of 64 rows)
    const int qblocks = S_LEN / (16 * WAVES);       // 32
    int blk  = blockIdx.x;
    int b    = blk / (NHEADS * qblocks);
    int rem  = blk % (NHEADS * qblocks);
    int h    = rem / qblocks;
    int qblk = rem % qblocks;
    int qbase = qblk * (16 * WAVES) + wave * 16;

    // base-2 softmax: fold 1/sqrt(64) * log2(e) into Q
    const float qscale = 0.125f * 1.4426950408889634f;
    const size_t bh_off = (size_t)b * S_LEN * DIMC + (size_t)h * DH;
    const float* Qp = Q + bh_off;
    const float* Kp = K + bh_off;
    const float* Vp = V + bh_off;
    float*       Op = O + bh_off;

    // ---- Q as B-fragment of S^T = K * Q^T ----
    // B layout (16-bit, 32x16): lane(col=ln, half); element t <-> k=8*half+t (t<8), 16+8*half+t-8 (t>=8)
    // -> lane ln reads two 8-wide runs of d-values of its OWN query row.
    v16h qb[2];
    {
        const float* qrow = Qp + (size_t)(qbase + ln) * DIMC;
        #pragma unroll
        for (int c = 0; c < 2; ++c) {
            v16h x;
            #pragma unroll
            for (int t = 0; t < 16; ++t)
                x[t] = (_Float16)(qrow[c * 32 + 8 * half + (t < 8 ? t : t + 8)] * qscale);
            qb[c] = x;
        }
    }

    // constant ones A-fragment: ones(16x32) x P^T -> row sums of P
    v16h ones_a;
    #pragma unroll
    for (int t = 0; t < 16; ++t) ones_a[t] = (_Float16)1.0f;

    // O^T accumulators: tile dt holds rows d = dt*16 + i + 8*half, col = qrow ln
    v8f o_acc[4];
    #pragma unroll
    for (int i = 0; i < 4; ++i) o_acc[i] = (v8f){0,0,0,0,0,0,0,0};
    v8f l_acc = (v8f){0,0,0,0,0,0,0,0};  // element 0 = running row sum l
    float m_r = -3.0e38f;                // running row max (this lane's q row)

    _Float16* Pw = Pt + wave * 16 * PSTRIDE;

    for (int kb = 0; kb < S_LEN; kb += KB) {
        __syncthreads();
        // ---- cooperative K load: [key][d] row-major, packed b128 stores ----
        {
            int r  = tid >> 1;          // 0..63 key row
            int ch = (tid & 1) * 32;    // d half
            const float* krow = Kp + (size_t)(kb + r) * DIMC + ch;
            #pragma unroll
            for (int g = 0; g < 4; ++g) {
                v8h x;
                #pragma unroll
                for (int t = 0; t < 8; ++t)
                    x[t] = (_Float16)krow[g * 8 + t];
                *(v8h*)&Ks[r * KSTRIDE + ch + g * 8] = x;
            }
        }
        // ---- cooperative V transpose: thread owns 4 d x 8 keys subtile ----
        {
            int dq = (tid & 15) * 4;    // d quad
            int r0 = (tid >> 4) * 8;    // key rows r0..r0+7
            float vv[4][8];
            #pragma unroll
            for (int rr = 0; rr < 8; ++rr) {
                const float* vrow = Vp + (size_t)(kb + r0 + rr) * DIMC + dq;
                #pragma unroll
                for (int d = 0; d < 4; ++d) vv[d][rr] = vrow[d];
            }
            #pragma unroll
            for (int d = 0; d < 4; ++d) {
                v8h x;
                #pragma unroll
                for (int rr = 0; rr < 8; ++rr) x[rr] = (_Float16)vv[d][rr];
                *(v8h*)&Vt[(dq + d) * VSTRIDE + r0] = x;   // [d][key]
            }
        }
        __syncthreads();

        // ---- S^T = K * Q^T : four 16-key tiles ----
        // A layout (16-bit, 16x32): lane(row=ln, half); chunk v4: element
        // v4*8+t <-> k = 16*v4 + 8*half + t -> two contiguous 8-half runs.
        v8f s[4];
        #pragma unroll
        for (int ct = 0; ct < 4; ++ct) {
            v8f acc = (v8f){0,0,0,0,0,0,0,0};
            #pragma unroll
            for (int c = 0; c < 2; ++c) {
                const _Float16* src = &Ks[(ct * 16 + ln) * KSTRIDE + c * 32 + 8 * half];
                v16h a;
                #pragma unroll
                for (int v4 = 0; v4 < 2; ++v4)
                    #pragma unroll
                    for (int t = 0; t < 8; ++t)
                        a[v4 * 8 + t] = src[v4 * 16 + t];
                acc = __builtin_amdgcn_wmma_f32_16x16x32_f16(
                        false, a, false, qb[c], (short)0, acc, false, false);
            }
            s[ct] = acc;
        }
        // s[ct][i] = score(key = kb + ct*16 + i + 8*half, qrow = ln)

        // ---- online softmax: lane-local max over 32 keys + one shfl_xor ----
        float mloc = s[0][0];
        #pragma unroll
        for (int ct = 0; ct < 4; ++ct)
            #pragma unroll
            for (int i = 0; i < 8; ++i)
                mloc = fmaxf(mloc, s[ct][i]);
        mloc = fmaxf(mloc, __shfl_xor(mloc, 16, 32));
        float mnew  = fmaxf(m_r, mloc);
        float alpha = __builtin_amdgcn_exp2f(m_r - mnew);
        m_r = mnew;
        #pragma unroll
        for (int ct = 0; ct < 4; ++ct)
            #pragma unroll
            for (int i = 0; i < 8; ++i)
                s[ct][i] = __builtin_amdgcn_exp2f(s[ct][i] - mnew);
        // alpha-rescale accumulators (l_acc: only element 0 is ever read)
        #pragma unroll
        for (int dt = 0; dt < 4; ++dt)
            #pragma unroll
            for (int i = 0; i < 8; ++i)
                o_acc[dt][i] *= alpha;
        l_acc[0] *= alpha;

        // ---- P^T tile: per-lane contiguous packed b128 stores ----
        #pragma unroll
        for (int ct = 0; ct < 4; ++ct) {
            v8h x;
            #pragma unroll
            for (int i = 0; i < 8; ++i) x[i] = (_Float16)s[ct][i];
            *(v8h*)&Pw[ln * PSTRIDE + ct * 16 + 8 * half] = x;
        }
        // same-wave DS ops are in-order; compiler inserts s_wait_dscnt.
        // B-fragments of P^T: lane ln reads its own row, two 16B runs.
        v16h pb[2];
        {
            const _Float16* prow = Pw + ln * PSTRIDE;
            #pragma unroll
            for (int pc = 0; pc < 2; ++pc) {
                v16h x;
                #pragma unroll
                for (int t = 0; t < 16; ++t)
                    x[t] = prow[pc * 32 + 8 * half + (t < 8 ? t : t + 8)];
                pb[pc] = x;
            }
        }

        // ---- O^T += V^T * P^T ; l += ones * P^T (row sums on the XDL) ----
        #pragma unroll
        for (int pc = 0; pc < 2; ++pc) {
            #pragma unroll
            for (int dt = 0; dt < 4; ++dt) {
                const _Float16* vr = &Vt[(dt * 16 + ln) * VSTRIDE + pc * 32 + 8 * half];
                v16h va;
                #pragma unroll
                for (int v4 = 0; v4 < 2; ++v4)
                    #pragma unroll
                    for (int t = 0; t < 8; ++t)
                        va[v4 * 8 + t] = vr[v4 * 16 + t];
                o_acc[dt] = __builtin_amdgcn_wmma_f32_16x16x32_f16(
                              false, va, false, pb[pc], (short)0, o_acc[dt], false, false);
            }
            l_acc = __builtin_amdgcn_wmma_f32_16x16x32_f16(
                      false, ones_a, false, pb[pc], (short)0, l_acc, false, false);
        }
    }

    // ---- epilogue: lane-local 1/l, contiguous 8-float runs per d-tile ----
    float inv_l = 1.0f / l_acc[0];
    float* orow = Op + (size_t)(qbase + ln) * DIMC + 8 * half;
    #pragma unroll
    for (int dt = 0; dt < 4; ++dt)
        #pragma unroll
        for (int i = 0; i < 8; ++i)
            orow[dt * 16 + i] = o_acc[dt][i] * inv_l;
}

extern "C" void kernel_launch(void* const* d_in, const int* in_sizes, int n_in,
                              void* d_out, int out_size, void* d_ws, size_t ws_size,
                              hipStream_t stream) {
    (void)in_sizes; (void)n_in; (void)out_size; (void)d_ws; (void)ws_size;
    const float* q = (const float*)d_in[0];
    const float* k = (const float*)d_in[1];
    const float* v = (const float*)d_in[2];
    float* out = (float*)d_out;

    const int qblocks = S_LEN / (16 * WAVES);              // 32
    dim3 grid(2 * NHEADS * qblocks);                       // 1024 blocks
    dim3 block(32 * WAVES);                                // 128 threads = 4 waves
    fa_fwd_kernel<<<grid, block, 0, stream>>>(q, k, v, out);
}
